// TGATPredictorAblation_34170759807353
// MI455X (gfx1250) — matrix operand order repaired and checked by reading, not compiled, not run on hardware
//
#include <hip/hip_runtime.h>
#include <math.h>

typedef __attribute__((ext_vector_type(16))) __bf16 v16bf;
typedef __attribute__((ext_vector_type(8)))  float  v8f;
typedef __attribute__((ext_vector_type(8)))  __bf16 bf16x8;

#define HH   256   // hidden
#define DD   64    // x feature dim
#define EE   32    // embedding dim
#define TT   8     // time-encoding freqs
#define PP   96    // D+E
#define KINC 272   // H + 2T
#define LMAX 200
#define NH   4
#define DH   64

// ---------------------------------------------------------------------------
// Prep: transpose 96x256 fp32 weight -> 256x96 bf16 (so WMMA B fragments are
// contiguous 16B loads). 24576 elements; trivial one-shot kernel.
// ---------------------------------------------------------------------------
__global__ __launch_bounds__(256)
void transpose_w_bf16(const float* __restrict__ W, void* __restrict__ Wt_v)
{
  __bf16* Wt = (__bf16*)Wt_v;
  int i = blockIdx.x * 256 + threadIdx.x;
  if (i < PP * HH) {
    int n = i / PP, k = i % PP;
    Wt[i] = (__bf16)W[(size_t)k * HH + n];
  }
}

// ---------------------------------------------------------------------------
// Input projection: out[M,256] = [x | emb[idx]] (M x 96) @ W (96 x 256) + b
// bf16 WMMA, fp32 accumulate. Block = 256 thr = 8 waves; wave w owns column
// tiles {w, w+8} (two accumulators) so one block covers all N=256 columns.
// A tile staged in LDS as bf16; B from pre-transposed bf16 weight (contiguous).
// ---------------------------------------------------------------------------
template<bool OUT_BF16>
__global__ __launch_bounds__(256)
void input_proj_wmma(const float* __restrict__ x,
                     const int*   __restrict__ idx,
                     const float* __restrict__ emb,
                     const void*  __restrict__ Wt_v,   // 256x96 bf16
                     const float* __restrict__ bias,
                     void* __restrict__ outp, int M)
{
  const __bf16* __restrict__ Wt = (const __bf16*)Wt_v;
  __shared__ __bf16 As[16 * PP];           // 16 rows x 96, bf16 (rows 192B, 16B-aligned)
  const int tid  = threadIdx.x;
  const int wave = tid >> 5;
  const int lane = tid & 31;
  const int row0 = blockIdx.x * 16;

  // Stage A-tile, fusing concat + embedding gather + bf16 convert.
  for (int p = tid; p < 16 * PP; p += 256) {
    int r = p / PP, c = p % PP;
    int grow = row0 + r;
    float v = 0.f;
    if (grow < M) {
      v = (c < DD) ? x[(size_t)grow * DD + c]
                   : emb[(size_t)idx[grow] * EE + (c - DD)];
    }
    As[r * PP + c] = (__bf16)v;
  }
  __syncthreads();

  const int m  = lane & 15;      // A row / B,C column-in-tile
  const int u  = lane >> 4;      // lane half selects K/M sub-block
  const int n0 = wave * 16 + m;
  const int n1 = 128 + wave * 16 + m;

  v8f acc0 = {0.f,0.f,0.f,0.f,0.f,0.f,0.f,0.f};
  v8f acc1 = {0.f,0.f,0.f,0.f,0.f,0.f,0.f,0.f};

  #pragma unroll
  for (int ks = 0; ks < 3; ++ks) {
    // A 16x32 bf16 fragment (ISA 7.12.2): lanes 0-15 K={0..7,16..23}, lanes 16-31 +8
    const int ka = ks * 32 + u * 8;
    bf16x8 alo = *(const bf16x8*)(As + m * PP + ka);
    bf16x8 ahi = *(const bf16x8*)(As + m * PP + ka + 16);
    v16bf a = __builtin_shufflevector(alo, ahi, 0,1,2,3,4,5,6,7,8,9,10,11,12,13,14,15);
    // B 32x16 bf16 fragment: same K order as A: lanes 0-15 K={0..7,16..23}, lanes 16-31 +8;
    // contiguous in the transposed weight.
    const int kb = ks * 32 + u * 8;
    {
      bf16x8 blo = *(const bf16x8*)(Wt + (size_t)n0 * PP + kb);
      bf16x8 bhi = *(const bf16x8*)(Wt + (size_t)n0 * PP + kb + 16);
      v16bf b = __builtin_shufflevector(blo, bhi, 0,1,2,3,4,5,6,7,8,9,10,11,12,13,14,15);
      acc0 = __builtin_amdgcn_wmma_f32_16x16x32_bf16(false, a, false, b,
                                                     (short)0, acc0, false, false);
    }
    {
      bf16x8 blo = *(const bf16x8*)(Wt + (size_t)n1 * PP + kb);
      bf16x8 bhi = *(const bf16x8*)(Wt + (size_t)n1 * PP + kb + 16);
      v16bf b = __builtin_shufflevector(blo, bhi, 0,1,2,3,4,5,6,7,8,9,10,11,12,13,14,15);
      acc1 = __builtin_amdgcn_wmma_f32_16x16x32_bf16(false, a, false, b,
                                                     (short)0, acc1, false, false);
    }
  }

  // C/D layout: VGPR r -> M = r (lanes 0-15) / r+8 (lanes 16-31), N = lane%16
  const float bn0 = bias[n0];
  const float bn1 = bias[n1];
  #pragma unroll
  for (int r = 0; r < 8; ++r) {
    int grow = row0 + u * 8 + r;
    if (grow < M) {
      float v0 = acc0[r] + bn0;
      float v1 = acc1[r] + bn1;
      if constexpr (OUT_BF16) {
        ((__bf16*)outp)[(size_t)grow * HH + n0] = (__bf16)v0;
        ((__bf16*)outp)[(size_t)grow * HH + n1] = (__bf16)v1;
      } else {
        ((float*)outp)[(size_t)grow * HH + n0] = v0;
        ((float*)outp)[(size_t)grow * HH + n1] = v1;
      }
    }
  }
}

// ---------------------------------------------------------------------------
// Block-wide sum reduce (256 threads)
// ---------------------------------------------------------------------------
__device__ __forceinline__ float block_sum(float v, float* red, int t) {
  red[t] = v; __syncthreads();
  for (int off = 128; off > 0; off >>= 1) {
    if (t < off) red[t] += red[t + off];
    __syncthreads();
  }
  float r = red[0];
  __syncthreads();
  return r;
}

// ---------------------------------------------------------------------------
// Fused TGAT layer, one block per batch element. Folded forms:
//   wk4[j][hd] = (Wk[:,hd] q_hd)/sqrt(Dh)      (272 x 4, head-interleaved)
//   logit      = kv_l . wk[hd]                  (K never materialized)
//   s4[j][hd]  = sum_l att[hd,l] kv_l[j]        (272 x 4)
//   agg[hd]    = s[hd] @ Wv[:,hd]               (V never materialized)
// then residual + LN + FFN + LN, h updated in place.
// Head-interleaved LDS layouts make every hot access one ds_load_b128.
// ---------------------------------------------------------------------------
__global__ __launch_bounds__(256)
void layer_kernel(float* __restrict__ h,
                  const void* __restrict__ ctx_v,
                  const float* __restrict__ t_ctx,
                  const int*  __restrict__ actc,
                  const float* __restrict__ freq, const float* __restrict__ phase,
                  const float* __restrict__ Wq,  const float* __restrict__ Wk,
                  const float* __restrict__ Wv,
                  const float* __restrict__ Wres, const float* __restrict__ bres,
                  const float* __restrict__ Wff1, const float* __restrict__ bff1,
                  const float* __restrict__ Wff2, const float* __restrict__ bff2,
                  const float* __restrict__ g1, const float* __restrict__ beta1,
                  const float* __restrict__ g2, const float* __restrict__ beta2,
                  int L)
{
  const __bf16* __restrict__ ctx = (const __bf16*)ctx_v;
  const int b = blockIdx.x;
  const int t = threadIdx.x;

  __shared__ float h_s[HH];
  __shared__ float q_s[HH];
  __shared__ __align__(16) float wk4[KINC * NH];   // [j][hd]
  __shared__ __align__(16) float s4[KINC * NH];    // [j][hd]
  __shared__ __align__(16) float att4[LMAX * NH];  // [l][hd]
  __shared__ float logt_s[LMAX];
  __shared__ float red[256];
  __shared__ float z1_s[HH];
  __shared__ float a1_s[HH];

  h_s[t] = h[(size_t)b * HH + t];
  __syncthreads();

  // q = h @ Wq (column reads coalesced across threads)
  {
    float acc = 0.f;
    for (int j = 0; j < HH; ++j) acc += h_s[j] * Wq[(size_t)j * HH + t];
    q_s[t] = acc;
  }
  __syncthreads();

  // wk4[j*4+hd] = (1/8) * sum_d Wk[j, hd*64+d] * q[hd*64+d]
  // p = j*4+hd so writes are contiguous; a wave covers whole Wk rows.
  for (int p = t; p < NH * KINC; p += 256) {
    int j = p >> 2, hd = p & 3;
    const float* wrow = Wk + (size_t)j * HH + hd * DH;
    const float* qh   = q_s + hd * DH;
    float acc = 0.f;
    #pragma unroll 8
    for (int d = 0; d < DH; ++d) acc += wrow[d] * qh[d];
    wk4[p] = acc * 0.125f;   // 1/sqrt(Dh), Dh=64
  }
  __syncthreads();

  // Attention logits: thread l computes kv_l . wk[hd] for all 4 heads.
  if (t < L) {
    const int l = t;
    float lt = log1pf(fmaxf(t_ctx[(size_t)b * L + l], 0.f));
    logt_s[l] = lt;
    float a0 = 0.f, a1 = 0.f, a2 = 0.f, a3 = 0.f;
    const __bf16* cr = ctx + ((size_t)b * L + l) * HH;
    for (int j8 = 0; j8 < HH; j8 += 8) {
      bf16x8 pk = *(const bf16x8*)(cr + j8);
      #pragma unroll
      for (int e = 0; e < 8; ++e) {
        float v = (float)pk[e];
        const float4 wv = *(const float4*)&wk4[(j8 + e) * 4];
        a0 += v * wv.x; a1 += v * wv.y; a2 += v * wv.z; a3 += v * wv.w;
      }
    }
    #pragma unroll
    for (int k = 0; k < TT; ++k) {
      float z = lt * freq[k] + phase[k];
      float sv = sinf(z), cv = cosf(z);
      const float4 ws_ = *(const float4*)&wk4[(HH + k) * 4];
      const float4 wc_ = *(const float4*)&wk4[(HH + TT + k) * 4];
      a0 += sv * ws_.x + cv * wc_.x;
      a1 += sv * ws_.y + cv * wc_.y;
      a2 += sv * ws_.z + cv * wc_.z;
      a3 += sv * ws_.w + cv * wc_.w;
    }
    bool msk = actc[(size_t)b * L + l] > 0;
    *(float4*)&att4[l * 4] = msk ? make_float4(a0, a1, a2, a3)
                                 : make_float4(-1e9f, -1e9f, -1e9f, -1e9f);
  }
  __syncthreads();

  // Softmax over L per head: 64 threads per head (head-strided LDS reads).
  {
    const int hd = t >> 6, g = t & 63;
    float mx = -3.0e38f;
    for (int l = g; l < L; l += 64) mx = fmaxf(mx, att4[l * 4 + hd]);
    red[t] = mx; __syncthreads();
    for (int off = 32; off > 0; off >>= 1) {
      if (g < off) red[t] = fmaxf(red[t], red[t + off]);
      __syncthreads();
    }
    mx = red[hd << 6]; __syncthreads();
    float sm = 0.f;
    for (int l = g; l < L; l += 64) {
      float e = expf(att4[l * 4 + hd] - mx);
      att4[l * 4 + hd] = e;
      sm += e;
    }
    red[t] = sm; __syncthreads();
    for (int off = 32; off > 0; off >>= 1) {
      if (g < off) red[t] += red[t + off];
      __syncthreads();
    }
    sm = red[hd << 6]; __syncthreads();
    float inv = 1.f / sm;
    for (int l = g; l < L; l += 64) att4[l * 4 + hd] *= inv;
  }
  __syncthreads();

  // s4[j][hd] = sum_l att[l][hd] * kv[l][j]; ctx part coalesced (thread = j).
  {
    float a0 = 0.f, a1 = 0.f, a2 = 0.f, a3 = 0.f;
    const __bf16* col = ctx + (size_t)b * L * HH + t;
    for (int l = 0; l < L; ++l) {
      if (l + 16 < L) __builtin_prefetch(col + (size_t)(l + 16) * HH, 0, 0);
      float v = (float)col[(size_t)l * HH];
      const float4 at = *(const float4*)&att4[l * 4];
      a0 += at.x * v; a1 += at.y * v; a2 += at.z * v; a3 += at.w * v;
    }
    *(float4*)&s4[t * 4] = make_float4(a0, a1, a2, a3);
  }
  if (t < 64) {  // time-encoding tail of kv
    int hd = t >> 4, k = t & 15;
    float acc = 0.f;
    for (int l = 0; l < L; ++l) {
      float lt = logt_s[l];
      float tv = (k < TT) ? sinf(lt * freq[k] + phase[k])
                          : cosf(lt * freq[k - TT] + phase[k - TT]);
      acc += att4[l * 4 + hd] * tv;
    }
    s4[(HH + k) * 4 + hd] = acc;
  }
  __syncthreads();

  // agg = s @ Wv[:, hd-slice], residual h @ Wres, LN1
  float r1;
  {
    const int hd = t >> 6;
    float agg = 0.f;
    for (int j = 0; j < KINC; ++j) agg += s4[j * 4 + hd] * Wv[(size_t)j * HH + t];
    float res = 0.f;
    for (int j = 0; j < HH; ++j) res += h_s[j] * Wres[(size_t)j * HH + t];
    r1 = agg + res + bres[t];
  }
  {
    float mu  = block_sum(r1, red, t) * (1.f / HH);
    float d   = r1 - mu;
    float var = block_sum(d * d, red, t) * (1.f / HH);
    z1_s[t] = d * rsqrtf(var + 1e-5f) * g1[t] + beta1[t];
  }
  __syncthreads();

  // FFN
  {
    float acc = 0.f;
    for (int j = 0; j < HH; ++j) acc += z1_s[j] * Wff1[(size_t)j * HH + t];
    a1_s[t] = fmaxf(acc + bff1[t], 0.f);
  }
  __syncthreads();
  float r2;
  {
    float acc = 0.f;
    for (int j = 0; j < HH; ++j) acc += a1_s[j] * Wff2[(size_t)j * HH + t];
    r2 = z1_s[t] + acc + bff2[t];
  }
  {
    float mu  = block_sum(r2, red, t) * (1.f / HH);
    float d   = r2 - mu;
    float var = block_sum(d * d, red, t) * (1.f / HH);
    h[(size_t)b * HH + t] = d * rsqrtf(var + 1e-5f) * g2[t] + beta2[t];
  }
}

// ---------------------------------------------------------------------------
// Readout head: z = relu(h W_ro + b); logits / sigmoid(dt) / sigmoid(rem)
// Output layout: [B*32 logits][B dt][B rem]
// ---------------------------------------------------------------------------
__global__ __launch_bounds__(256)
void readout_kernel(const float* __restrict__ h,
                    const float* __restrict__ W_ro,  const float* __restrict__ b_ro,
                    const float* __restrict__ W_cls, const float* __restrict__ b_cls,
                    const float* __restrict__ W_dt,  const float* __restrict__ b_dt,
                    const float* __restrict__ W_rem, const float* __restrict__ b_rem,
                    float* __restrict__ out, int B)
{
  const int b = blockIdx.x, t = threadIdx.x;
  __shared__ float hs[HH], zs[HH];
  hs[t] = h[(size_t)b * HH + t];
  __syncthreads();
  float acc = 0.f;
  for (int j = 0; j < HH; ++j) acc += hs[j] * W_ro[(size_t)j * HH + t];
  zs[t] = fmaxf(acc + b_ro[t], 0.f);
  __syncthreads();
  if (t < 32) {
    float a = 0.f;
    for (int j = 0; j < HH; ++j) a += zs[j] * W_cls[(size_t)j * 32 + t];
    out[(size_t)b * 32 + t] = a + b_cls[t];
  } else if (t == 32) {
    float a = 0.f;
    for (int j = 0; j < HH; ++j) a += zs[j] * W_dt[j];
    out[(size_t)B * 32 + b] = 1.f / (1.f + expf(-(a + b_dt[0])));
  } else if (t == 33) {
    float a = 0.f;
    for (int j = 0; j < HH; ++j) a += zs[j] * W_rem[j];
    out[(size_t)B * 33 + b] = 1.f / (1.f + expf(-(a + b_rem[0])));
  }
}

// ---------------------------------------------------------------------------
extern "C" void kernel_launch(void* const* d_in, const int* in_sizes, int n_in,
                              void* d_out, int out_size, void* d_ws, size_t ws_size,
                              hipStream_t stream) {
  const float* x_u      = (const float*)d_in[0];
  const float* x_ctx    = (const float*)d_in[1];
  const float* t_ctx    = (const float*)d_in[2];
  const int*   act_u    = (const int*)  d_in[3];
  const int*   act_c    = (const int*)  d_in[4];
  const float* emb      = (const float*)d_in[5];
  const float* W_in_self= (const float*)d_in[6];
  const float* b_in_self= (const float*)d_in[7];
  const float* W_in_ctx = (const float*)d_in[8];
  const float* b_in_ctx = (const float*)d_in[9];
  const float* freq     = (const float*)d_in[10];
  const float* phase    = (const float*)d_in[11];
  const float* Wq       = (const float*)d_in[12];
  const float* Wk       = (const float*)d_in[13];
  const float* Wv       = (const float*)d_in[14];
  const float* Wres     = (const float*)d_in[15];
  const float* bres     = (const float*)d_in[16];
  const float* Wff1     = (const float*)d_in[17];
  const float* bff1     = (const float*)d_in[18];
  const float* Wff2     = (const float*)d_in[19];
  const float* bff2     = (const float*)d_in[20];
  const float* g1       = (const float*)d_in[21];
  const float* beta1    = (const float*)d_in[22];
  const float* g2       = (const float*)d_in[23];
  const float* beta2    = (const float*)d_in[24];
  const float* W_ro     = (const float*)d_in[25];
  const float* b_ro     = (const float*)d_in[26];
  const float* W_cls    = (const float*)d_in[27];
  const float* b_cls    = (const float*)d_in[28];
  const float* W_dt     = (const float*)d_in[29];
  const float* b_dt     = (const float*)d_in[30];
  const float* W_rem    = (const float*)d_in[31];
  const float* b_rem    = (const float*)d_in[32];

  const int B  = in_sizes[0] / DD;          // 1024
  const int L  = in_sizes[1] / (B * DD);    // 200
  const int NL = in_sizes[10] / TT;         // 3
  const int M  = B * L;

  // Workspace: [Wt_self bf16 48K][Wt_ctx bf16 48K][ctx bf16][h fp32]
  char* ws = (char*)d_ws;
  void* WtS = (void*)ws;
  void* WtC = (void*)(ws + (size_t)PP * HH * 2);
  char* base = ws + 2 * (size_t)PP * HH * 2;
  void* ctx = (void*)base;
  size_t ctx_bytes = ((size_t)M * HH * 2 + 255) & ~(size_t)255;
  float* h = (float*)(base + ctx_bytes);

  // 0) Transpose + bf16-convert the two input-projection weights.
  transpose_w_bf16<<<(PP * HH + 255) / 256, 256, 0, stream>>>(W_in_self, WtS);
  transpose_w_bf16<<<(PP * HH + 255) / 256, 256, 0, stream>>>(W_in_ctx,  WtC);

  // 1) h = [x_u | emb[act_u]] @ W_in_self + b   (fp32 out)
  input_proj_wmma<false><<<dim3((B + 15) / 16), 256, 0, stream>>>(
      x_u, act_u, emb, WtS, b_in_self, (void*)h, B);

  // 2) ctx = [x_ctx | emb[act_ctx]] @ W_in_ctx + b   (bf16 out)
  input_proj_wmma<true><<<dim3((M + 15) / 16), 256, 0, stream>>>(
      x_ctx, act_c, emb, WtC, b_in_ctx, ctx, M);

  // 3) NL fused attention layers (one block per batch element)
  for (int i = 0; i < NL; ++i) {
    layer_kernel<<<B, 256, 0, stream>>>(
        h, ctx, t_ctx, act_c,
        freq + (size_t)i * TT, phase + (size_t)i * TT,
        Wq   + (size_t)i * HH * HH,
        Wk   + (size_t)i * KINC * HH,
        Wv   + (size_t)i * KINC * HH,
        Wres + (size_t)i * HH * HH, bres + (size_t)i * HH,
        Wff1 + (size_t)i * HH * HH, bff1 + (size_t)i * HH,
        Wff2 + (size_t)i * HH * HH, bff2 + (size_t)i * HH,
        g1 + (size_t)i * HH, beta1 + (size_t)i * HH,
        g2 + (size_t)i * HH, beta2 + (size_t)i * HH, L);
  }

  // 4) Readout heads
  readout_kernel<<<B, 256, 0, stream>>>(
      h, W_ro, b_ro, W_cls, b_cls, W_dt, b_dt, W_rem, b_rem,
      (float*)d_out, B);
}
